// MultiHeadAttention_51986284150861
// MI455X (gfx1250) — matrix operand order repaired and checked by reading, not compiled, not run on hardware
//
#include <hip/hip_runtime.h>
#include <hip/hip_bf16.h>
#include <math.h>

// ---------------------------------------------------------------------------
// Transformer encoder block for MI455X (gfx1250, wave32, WMMA).
//   H=1024, NH=16, D=64, B=2, T=2048  -> M rows = 4096 everywhere.
// All matmuls run on v_wmma_f32_16x16x32_bf16 (fp32 -> bf16 operands,
// fp32 accum). Attention is single-pass flash style (no 2048x2048 logits
// in HBM). GEMM K-loop is double-buffered in LDS (one barrier per K-step,
// next tile prefetched to registers while WMMAs run on the current tile).
// LDS tiles padded (stride 40 bf16 = 20 banks) -> conflict-free b128 reads.
// ---------------------------------------------------------------------------

#define H_DIM 1024
#define NHEAD 16
#define HD 64
#define TSEQ 2048
#define BATCH 2
#define MROWS (BATCH * TSEQ)   // 4096

typedef __attribute__((ext_vector_type(16))) __bf16 v16bf;
typedef __attribute__((ext_vector_type(8)))  __bf16 v8bf;
typedef __attribute__((ext_vector_type(4)))  __bf16 v4bf;
typedef __attribute__((ext_vector_type(8)))  float  v8f;

__device__ __forceinline__ __bf16 f2bf(float f) {
  unsigned u = __builtin_bit_cast(unsigned, f);
  u += 0x7fffu + ((u >> 16) & 1u);            // round-to-nearest-even
  unsigned short h = (unsigned short)(u >> 16);
  return __builtin_bit_cast(__bf16, h);
}

__device__ __forceinline__ v16bf combine8(v8bf lo, v8bf hi) {
  v16bf r;
#pragma unroll
  for (int i = 0; i < 8; ++i) { r[i] = lo[i]; r[i + 8] = hi[i]; }
  return r;
}

// ---------------------------------------------------------------------------
// Generic WMMA GEMM:  C[M,N] = A[M,K] @ W[N,K]^T  (+ epilogue by MODE)
// block = 256 threads (8 waves), C tile 128x128, wave grid 2(M) x 4(N),
// each wave: 4(M) x 2(N) 16x16 accumulators. K step = 32. Double-buffered.
// ---------------------------------------------------------------------------
#define LDSS 40   // bf16 elements per LDS tile row (32 + 8 pad)

enum GemmMode { MODE_QK_HEADS = 0, MODE_V_HEADS_T = 1, MODE_F32_RES = 2,
                MODE_F32_BIAS_GELU = 3, MODE_F32_BIAS_RES = 4 };

template <int MODE>
__global__ __launch_bounds__(256)
void gemm_bf16_wmma(const float* __restrict__ A,    // [M,K]
                    const float* __restrict__ W,    // [N,K]
                    const float* __restrict__ bias, // [N] or null
                    const float* __restrict__ res,  // [M,N] or null
                    void* __restrict__ outp,
                    int M, int N, int K)
{
  __shared__ __align__(16) __bf16 sA[2][128 * LDSS];
  __shared__ __align__(16) __bf16 sB[2][128 * LDSS];

  const int tid  = threadIdx.x;
  const int lane = tid & 31;
  const int wave = tid >> 5;
  const int wm   = wave >> 2;        // 0..1, 64 M-rows each
  const int wn   = wave & 3;         // 0..3, 32 N-cols each
  const int l16  = lane & 15;
  const bool hih = lane >= 16;

  const int bm = blockIdx.y * 128;
  const int bn = blockIdx.x * 128;

  // staging slots: this thread owns 4 float4 of A and 4 of W per K-tile
  const int srow = tid >> 1;            // unused helper for prefetch only
  float4 ra[4], rb[4];

  v8f acc[4][2];
#pragma unroll
  for (int i = 0; i < 4; ++i)
#pragma unroll
    for (int j = 0; j < 2; ++j)
#pragma unroll
      for (int e = 0; e < 8; ++e) acc[i][j][e] = 0.0f;

  // ---- prologue: fetch K-tile 0 into registers, stage to LDS buffer 0 ----
#pragma unroll
  for (int q = 0; q < 4; ++q) {
    int f4  = tid + 256 * q;            // 0..1023 float4 slots
    int row = f4 >> 3;
    int c4  = f4 & 7;
    ra[q] = *(const float4*)(A + (size_t)(bm + row) * K + c4 * 4);
    rb[q] = *(const float4*)(W + (size_t)(bn + row) * K + c4 * 4);
  }
  int cur = 0;
#pragma unroll
  for (int q = 0; q < 4; ++q) {
    int f4  = tid + 256 * q;
    int row = f4 >> 3;
    int c4  = f4 & 7;
    v4bf ba = { f2bf(ra[q].x), f2bf(ra[q].y), f2bf(ra[q].z), f2bf(ra[q].w) };
    *(v4bf*)&sA[cur][row * LDSS + c4 * 4] = ba;
    v4bf bb = { f2bf(rb[q].x), f2bf(rb[q].y), f2bf(rb[q].z), f2bf(rb[q].w) };
    *(v4bf*)&sB[cur][row * LDSS + c4 * 4] = bb;
  }
  __syncthreads();

  for (int k0 = 0; k0 < K; k0 += 32) {
    const bool more = (k0 + 32) < K;
    // ---- prefetch next K-tile into registers (overlaps with WMMA below) ---
    if (more) {
#pragma unroll
      for (int q = 0; q < 4; ++q) {
        int f4  = tid + 256 * q;
        int row = f4 >> 3;
        int c4  = f4 & 7;
        ra[q] = *(const float4*)(A + (size_t)(bm + row) * K + k0 + 32 + c4 * 4);
        rb[q] = *(const float4*)(W + (size_t)(bn + row) * K + k0 + 32 + c4 * 4);
      }
      __builtin_prefetch(A + (size_t)(bm + srow) * K + k0 + 64 + (tid & 1) * 16, 0, 1);
      __builtin_prefetch(W + (size_t)(bn + srow) * K + k0 + 64 + (tid & 1) * 16, 0, 1);
    }

    // ---- compute on current buffer -----------------------------------
    // A fragments: lanes 0-15 take K{0..7,16..23}, lanes 16-31 K{8..15,24..31}
    v16bf af[4];
#pragma unroll
    for (int i = 0; i < 4; ++i) {
      int r = wm * 64 + i * 16 + l16;
      const __bf16* base = &sA[cur][r * LDSS + (hih ? 8 : 0)];
      af[i] = combine8(*(const v8bf*)base, *(const v8bf*)(base + 16));
    }
    // B fragments: lanes 0-15 take K{0..7,16..23}, lanes 16-31 K{8..15,24..31}
    v16bf bfv[2];
#pragma unroll
    for (int j = 0; j < 2; ++j) {
      int n = wn * 32 + j * 16 + l16;
      const __bf16* base = &sB[cur][n * LDSS + (hih ? 8 : 0)];
      bfv[j] = combine8(*(const v8bf*)base, *(const v8bf*)(base + 16));
    }
#pragma unroll
    for (int i = 0; i < 4; ++i)
#pragma unroll
      for (int j = 0; j < 2; ++j)
        acc[i][j] = __builtin_amdgcn_wmma_f32_16x16x32_bf16(
            false, af[i], false, bfv[j], (short)0, acc[i][j], false, false);

    // ---- publish next tile to the other buffer; one barrier per step ----
    if (more) {
      int nxt = cur ^ 1;
#pragma unroll
      for (int q = 0; q < 4; ++q) {
        int f4  = tid + 256 * q;
        int row = f4 >> 3;
        int c4  = f4 & 7;
        v4bf ba = { f2bf(ra[q].x), f2bf(ra[q].y), f2bf(ra[q].z), f2bf(ra[q].w) };
        *(v4bf*)&sA[nxt][row * LDSS + c4 * 4] = ba;
        v4bf bb = { f2bf(rb[q].x), f2bf(rb[q].y), f2bf(rb[q].z), f2bf(rb[q].w) };
        *(v4bf*)&sB[nxt][row * LDSS + c4 * 4] = bb;
      }
      __syncthreads();
      cur = nxt;
    }
  }

  // epilogue: C layout -> lane l16 = N col, vgpr r = M row (+8 for hi half)
#pragma unroll
  for (int i = 0; i < 4; ++i) {
#pragma unroll
    for (int j = 0; j < 2; ++j) {
#pragma unroll
      for (int r = 0; r < 8; ++r) {
        int m = bm + wm * 64 + i * 16 + r + (hih ? 8 : 0);
        int n = bn + wn * 32 + j * 16 + l16;
        float c = acc[i][j][r];
        if (MODE == MODE_QK_HEADS || MODE == MODE_V_HEADS_T) {
          int b = m >> 11, t = m & (TSEQ - 1);
          int hd = n >> 6, d = n & (HD - 1);
          __bf16* ob = (__bf16*)outp;
          size_t idx = (MODE == MODE_QK_HEADS)
              ? (((size_t)(b * NHEAD + hd) * TSEQ + t) * HD + d)     // [B,NH,T,D]
              : (((size_t)(b * NHEAD + hd) * HD + d) * TSEQ + t);    // [B,NH,D,T]
          ob[idx] = f2bf(c);
        } else {
          float* of = (float*)outp;
          size_t idx = (size_t)m * N + n;
          if (MODE == MODE_F32_RES) {
            of[idx] = c + res[idx];
          } else if (MODE == MODE_F32_BIAS_GELU) {
            float x = c + bias[n];
            of[idx] = 0.5f * x * (1.0f + erff(x * 0.70710678118654752f));
          } else { // MODE_F32_BIAS_RES
            of[idx] = c + bias[n] + res[idx];
          }
        }
      }
    }
  }
}

// ---------------------------------------------------------------------------
// Flash attention: grid (T/64, B*NH), block 128 (4 waves, 16 Q-rows each).
// Q:[B*NH,T,D] bf16, K:[B*NH,T,D] bf16, Vt:[B*NH,D,T] bf16 -> out fp32 [B*T,H]
// ---------------------------------------------------------------------------
__global__ __launch_bounds__(128)
void flash_attn(const __bf16* __restrict__ Q, const __bf16* __restrict__ Kb,
                const __bf16* __restrict__ Vt, const int* __restrict__ pm,
                float* __restrict__ out)
{
  __shared__ __align__(16) __bf16 Plds[4][16][72];  // per-wave P staging

  const int tid  = threadIdx.x;
  const int lane = tid & 31;
  const int wave = tid >> 5;
  const int l16  = lane & 15;
  const bool hih = lane >= 16;

  const int bh = blockIdx.y;          // 0..31
  const int b  = bh >> 4;
  const int h  = bh & 15;
  const int qt0 = blockIdx.x * 64 + wave * 16;

  const __bf16* Qh = Q  + (size_t)bh * TSEQ * HD;
  const __bf16* Kh = Kb + (size_t)bh * TSEQ * HD;
  const __bf16* Vh = Vt + (size_t)bh * HD * TSEQ;
  const int* pmb = pm + b * TSEQ;

  // Q A-fragments over D=64 (2 K-chunks of 32), held in registers for all tk
  v16bf qf[2];
#pragma unroll
  for (int kc = 0; kc < 2; ++kc) {
    const __bf16* base = Qh + (size_t)(qt0 + l16) * HD + kc * 32 + (hih ? 8 : 0);
    qf[kc] = combine8(*(const v8bf*)base, *(const v8bf*)(base + 16));
  }

  const float scale = 0.125f;  // 1/sqrt(64)
  v8f o[4];
#pragma unroll
  for (int j = 0; j < 4; ++j)
#pragma unroll
    for (int e = 0; e < 8; ++e) o[j][e] = 0.0f;
  float mrow[8], lrow[8];
#pragma unroll
  for (int r = 0; r < 8; ++r) { mrow[r] = -1e30f; lrow[r] = 0.0f; }

  for (int tk = 0; tk < TSEQ; tk += 64) {
    // S = Q K^T for 4 N-tiles (64 keys)
    v8f s[4];
    bool km[4];
#pragma unroll
    for (int j = 0; j < 4; ++j) {
#pragma unroll
      for (int e = 0; e < 8; ++e) s[j][e] = 0.0f;
      int t = tk + j * 16 + l16;
      km[j] = pmb[t] != 0;
#pragma unroll
      for (int kc = 0; kc < 2; ++kc) {
        const __bf16* base = Kh + (size_t)t * HD + kc * 32 + (hih ? 8 : 0);
        v16bf kf = combine8(*(const v8bf*)base, *(const v8bf*)(base + 16));
        s[j] = __builtin_amdgcn_wmma_f32_16x16x32_bf16(
            false, qf[kc], false, kf, (short)0, s[j], false, false);
      }
    }
    // online softmax (row stats shared across 16-lane half via shfl_xor)
    float mnew[8], alpha[8];
#pragma unroll
    for (int r = 0; r < 8; ++r) {
      float mx = -1e30f;
#pragma unroll
      for (int j = 0; j < 4; ++j) {
        float sv = km[j] ? s[j][r] * scale : -1e30f;
        mx = fmaxf(mx, sv);
      }
#pragma unroll
      for (int d = 1; d < 16; d <<= 1) mx = fmaxf(mx, __shfl_xor(mx, d, 32));
      mnew[r]  = fmaxf(mrow[r], mx);
      alpha[r] = __expf(mrow[r] - mnew[r]);
      mrow[r]  = mnew[r];
    }
#pragma unroll
    for (int r = 0; r < 8; ++r) {
      float rs = 0.0f;
      int prow = r + (hih ? 8 : 0);
#pragma unroll
      for (int j = 0; j < 4; ++j) {
        float p = km[j] ? __expf(s[j][r] * scale - mnew[r]) : 0.0f;
        rs += p;
        Plds[wave][prow][j * 16 + l16] = f2bf(p);   // C-layout -> LDS
      }
#pragma unroll
      for (int d = 1; d < 16; d <<= 1) rs += __shfl_xor(rs, d, 32);
      lrow[r] = lrow[r] * alpha[r] + rs;
#pragma unroll
      for (int j2 = 0; j2 < 4; ++j2) o[j2][r] *= alpha[r];
    }
    // O += P @ V   (P re-read from wave-private LDS in A-fragment layout)
#pragma unroll
    for (int kc = 0; kc < 2; ++kc) {
      const __bf16* pb = &Plds[wave][l16][kc * 32 + (hih ? 8 : 0)];
      v16bf pf = combine8(*(const v8bf*)pb, *(const v8bf*)(pb + 16));
#pragma unroll
      for (int j2 = 0; j2 < 4; ++j2) {
        int d = j2 * 16 + l16;
        const __bf16* vb = Vh + (size_t)d * TSEQ + tk + kc * 32 + (hih ? 8 : 0);
        v16bf vf = combine8(*(const v8bf*)vb, *(const v8bf*)(vb + 16));
        o[j2] = __builtin_amdgcn_wmma_f32_16x16x32_bf16(
            false, pf, false, vf, (short)0, o[j2], false, false);
      }
    }
  }
  // epilogue: 1/l normalization + post-softmax query mask
#pragma unroll
  for (int r = 0; r < 8; ++r) {
    int t = qt0 + r + (hih ? 8 : 0);
    bool qm = pmb[t] != 0;
    float inv = (qm && lrow[r] > 0.0f) ? 1.0f / lrow[r] : 0.0f;
#pragma unroll
    for (int j2 = 0; j2 < 4; ++j2) {
      int d = j2 * 16 + l16;
      out[(size_t)(b * TSEQ + t) * H_DIM + h * HD + d] = o[j2][r] * inv;
    }
  }
}

// ---------------------------------------------------------------------------
// Row LayerNorm over H=1024, one block (256 threads) per row.
// ---------------------------------------------------------------------------
__global__ __launch_bounds__(256)
void layernorm_kernel(const float* __restrict__ x, const float* __restrict__ g,
                      const float* __restrict__ bb, float* __restrict__ out)
{
  const int row = blockIdx.x;
  const int tid = threadIdx.x;
  const float4 v = *(const float4*)(x + (size_t)row * H_DIM + tid * 4);
  float s  = v.x + v.y + v.z + v.w;
  float s2 = v.x * v.x + v.y * v.y + v.z * v.z + v.w * v.w;
#pragma unroll
  for (int d = 1; d < 32; d <<= 1) {
    s += __shfl_xor(s, d, 32);
    s2 += __shfl_xor(s2, d, 32);
  }
  __shared__ float ws[8], ws2[8];
  if ((tid & 31) == 0) { ws[tid >> 5] = s; ws2[tid >> 5] = s2; }
  __syncthreads();
  float ts = 0.0f, ts2 = 0.0f;
#pragma unroll
  for (int i = 0; i < 8; ++i) { ts += ws[i]; ts2 += ws2[i]; }
  float mu  = ts * (1.0f / 1024.0f);
  float var = ts2 * (1.0f / 1024.0f) - mu * mu;
  float rs  = rsqrtf(var + 1e-5f);
  int c = tid * 4;
  float4 gv = *(const float4*)(g + c);
  float4 bv = *(const float4*)(bb + c);
  float4 ov;
  ov.x = (v.x - mu) * rs * gv.x + bv.x;
  ov.y = (v.y - mu) * rs * gv.y + bv.y;
  ov.z = (v.z - mu) * rs * gv.z + bv.z;
  ov.w = (v.w - mu) * rs * gv.w + bv.w;
  *(float4*)(out + (size_t)row * H_DIM + c) = ov;
}

// ---------------------------------------------------------------------------
extern "C" void kernel_launch(void* const* d_in, const int* in_sizes, int n_in,
                              void* d_out, int out_size, void* d_ws, size_t ws_size,
                              hipStream_t stream)
{
  const float* q   = (const float*)d_in[0];
  const float* k   = (const float*)d_in[1];
  const float* v   = (const float*)d_in[2];
  const int*   pm  = (const int*)  d_in[3];
  const float* Wq  = (const float*)d_in[4];
  const float* Wk  = (const float*)d_in[5];
  const float* Wv  = (const float*)d_in[6];
  const float* Wo  = (const float*)d_in[7];
  const float* W1  = (const float*)d_in[8];
  const float* b1  = (const float*)d_in[9];
  const float* W2  = (const float*)d_in[10];
  const float* b2  = (const float*)d_in[11];
  const float* g1  = (const float*)d_in[12];
  const float* be1 = (const float*)d_in[13];
  const float* g2  = (const float*)d_in[14];
  const float* be2 = (const float*)d_in[15];
  float* out = (float*)d_out;
  (void)in_sizes; (void)n_in; (void)out_size; (void)ws_size;

  char* ws = (char*)d_ws;
  __bf16* Qbf  = (__bf16*)(ws);                        //  8 MB  [B,NH,T,D]
  __bf16* Kbf  = (__bf16*)(ws + ((size_t)8  << 20));   //  8 MB  [B,NH,T,D]
  __bf16* Vtbf = (__bf16*)(ws + ((size_t)16 << 20));   //  8 MB  [B,NH,D,T]
  float*  attn = (float*) (ws + ((size_t)24 << 20));   // 16 MB  [4096,1024]
  float*  pre1 = (float*) (ws + ((size_t)40 << 20));   // 16 MB
  float*  h1   = (float*) (ws + ((size_t)56 << 20));   // 16 MB
  float*  ff1  = (float*) (ws + ((size_t)72 << 20));   // 64 MB  [4096,4096]
  float*  pre2 = attn;                                 // reuse (attn is dead)

  dim3 blk(256);
  dim3 gN1024(H_DIM / 128, MROWS / 128);   // (8, 32)
  dim3 gN4096(4096 / 128, MROWS / 128);    // (32, 32)

  gemm_bf16_wmma<MODE_QK_HEADS><<<gN1024, blk, 0, stream>>>(
      q, Wq, nullptr, nullptr, (void*)Qbf, MROWS, H_DIM, H_DIM);
  gemm_bf16_wmma<MODE_QK_HEADS><<<gN1024, blk, 0, stream>>>(
      k, Wk, nullptr, nullptr, (void*)Kbf, MROWS, H_DIM, H_DIM);
  gemm_bf16_wmma<MODE_V_HEADS_T><<<gN1024, blk, 0, stream>>>(
      v, Wv, nullptr, nullptr, (void*)Vtbf, MROWS, H_DIM, H_DIM);

  flash_attn<<<dim3(TSEQ / 64, BATCH * NHEAD), dim3(128), 0, stream>>>(
      Qbf, Kbf, Vtbf, pm, attn);

  gemm_bf16_wmma<MODE_F32_RES><<<gN1024, blk, 0, stream>>>(
      attn, Wo, nullptr, q, (void*)pre1, MROWS, H_DIM, H_DIM);
  layernorm_kernel<<<dim3(MROWS), blk, 0, stream>>>(pre1, g1, be1, h1);

  gemm_bf16_wmma<MODE_F32_BIAS_GELU><<<gN4096, blk, 0, stream>>>(
      h1, W1, b1, nullptr, (void*)ff1, MROWS, 4096, H_DIM);
  gemm_bf16_wmma<MODE_F32_BIAS_RES><<<gN1024, blk, 0, stream>>>(
      ff1, W2, b2, h1, (void*)pre2, MROWS, H_DIM, 4096);
  layernorm_kernel<<<dim3(MROWS), blk, 0, stream>>>(pre2, g2, be2, out);
}
